// MultiModalTransformer_16544214024881
// MI455X (gfx1250) — matrix operand order repaired and checked by reading, not compiled, not run on hardware
//
#include <hip/hip_runtime.h>

#define DEV __device__ __forceinline__

// ---------------------------------------------------------------- types
typedef __bf16        v16bf __attribute__((ext_vector_type(16)));
typedef float         v8f   __attribute__((ext_vector_type(8)));
typedef unsigned int  u32x4 __attribute__((ext_vector_type(4)));

union FragBF {
    u32x4          u[2];   // 32 bytes
    v16bf          v;
    unsigned short h[16];
};

// ---------------------------------------------------------------- config
constexpr int D  = 1024;
constexpr int L  = 6;
constexpr int H  = 16;
constexpr int F  = 4096;
constexpr int B  = 2;
constexpr int S  = 1024;
constexpr int DK = 64;
constexpr int T  = B * S;          // 2048 tokens
constexpr float EPS = 1e-5f;

// ---------------------------------------------------------------- helpers
DEV unsigned short f32_to_bf16(float f) {
    union { float f; unsigned u; } v; v.f = f;
    unsigned r = v.u + 0x7FFFu + ((v.u >> 16) & 1u);   // round-to-nearest-even
    return (unsigned short)(r >> 16);
}
DEV float bf16_to_f32(unsigned short h) {
    union { unsigned u; float f; } v; v.u = ((unsigned)h) << 16;
    return v.f;
}
DEV v8f wmma_bf16(const FragBF& a, const FragBF& b, v8f c) {
    return __builtin_amdgcn_wmma_f32_16x16x32_bf16(
        /*neg_a=*/false, a.v, /*neg_b=*/false, b.v,
        /*c_mod=*/(short)0, c, /*reuse_a=*/false, /*reuse_b=*/false);
}
// CDNA5 async DMA: global -> LDS, 16B per lane, tracked by ASYNCcnt.
DEV void async_ld_lds_b128(unsigned lds_off, const unsigned short* gptr) {
    asm volatile("global_load_async_to_lds_b128 %0, %1, off"
                 :: "v"(lds_off), "v"(gptr) : "memory");
}
DEV void wait_async_le1() { asm volatile("s_wait_asynccnt 0x1" ::: "memory"); }
DEV void wait_async_le0() { asm volatile("s_wait_asynccnt 0x0" ::: "memory"); }

// ---------------------------------------------------------------- weight convert
__global__ void cvt_f32_bf16(const float* __restrict__ in,
                             unsigned short* __restrict__ out, size_t n) {
    size_t i      = (size_t)blockIdx.x * blockDim.x + threadIdx.x;
    size_t stride = (size_t)gridDim.x * blockDim.x;
    for (; i < n; i += stride) out[i] = f32_to_bf16(in[i]);
}

// ---------------------------------------------------------------- embedding
__global__ void embed_kernel(const int* __restrict__ ids,
                             const float* __restrict__ tok,
                             const float* __restrict__ pos,
                             float* __restrict__ x) {
    int t = blockIdx.x;               // token 0..T-1
    int s = t & (S - 1);
    int id = ids[t];
    for (int d = threadIdx.x; d < D; d += blockDim.x)
        x[(size_t)t * D + d] = tok[(size_t)id * D + d] + pos[(size_t)s * D + d];
}

// ---------------------------------------------------------------- layernorm -> bf16
__global__ void ln_bf16_kernel(const float* __restrict__ x,
                               const float* __restrict__ g,
                               const float* __restrict__ b,
                               unsigned short* __restrict__ out) {
    __shared__ float red[256];
    int row = blockIdx.x;
    const float* xr = x + (size_t)row * D;
    float s = 0.f, s2 = 0.f;
    for (int d = threadIdx.x; d < D; d += 256) { float v = xr[d]; s += v; s2 += v * v; }
    red[threadIdx.x] = s; __syncthreads();
    for (int st = 128; st > 0; st >>= 1) {
        if (threadIdx.x < st) red[threadIdx.x] += red[threadIdx.x + st];
        __syncthreads();
    }
    float mu = red[0] / (float)D; __syncthreads();
    red[threadIdx.x] = s2; __syncthreads();
    for (int st = 128; st > 0; st >>= 1) {
        if (threadIdx.x < st) red[threadIdx.x] += red[threadIdx.x + st];
        __syncthreads();
    }
    float var  = red[0] / (float)D - mu * mu;
    float rstd = rsqrtf(var + EPS);
    for (int d = threadIdx.x; d < D; d += 256)
        out[(size_t)row * D + d] = f32_to_bf16((xr[d] - mu) * rstd * g[d] + b[d]);
}

// ---------------------------------------------------------------- final layernorm -> f32 (last token per batch)
__global__ void ln_final_kernel(const float* __restrict__ x,
                                const float* __restrict__ g,
                                const float* __restrict__ b,
                                float* __restrict__ out) {
    __shared__ float red[256];
    int row = blockIdx.x * S + (S - 1);
    const float* xr = x + (size_t)row * D;
    float s = 0.f, s2 = 0.f;
    for (int d = threadIdx.x; d < D; d += 256) { float v = xr[d]; s += v; s2 += v * v; }
    red[threadIdx.x] = s; __syncthreads();
    for (int st = 128; st > 0; st >>= 1) {
        if (threadIdx.x < st) red[threadIdx.x] += red[threadIdx.x + st];
        __syncthreads();
    }
    float mu = red[0] / (float)D; __syncthreads();
    red[threadIdx.x] = s2; __syncthreads();
    for (int st = 128; st > 0; st >>= 1) {
        if (threadIdx.x < st) red[threadIdx.x] += red[threadIdx.x + st];
        __syncthreads();
    }
    float var  = red[0] / (float)D - mu * mu;
    float rstd = rsqrtf(var + EPS);
    for (int d = threadIdx.x; d < D; d += 256)
        out[(size_t)blockIdx.x * D + d] = (xr[d] - mu) * rstd * g[d] + b[d];
}

// ---------------------------------------------------------------- WMMA GEMM: C[M,N] = A[M,K] @ W[N,K]^T
// Block = 256 threads (8 waves). Block tile = 128(M) x 64(N).
// W tile (64 x 32 halves) staged in LDS via async DMA, double buffered;
// each wave computes 16(M) x 64(N), reusing one A-frag across 4 B-frags.
// flags bit0: add residual (float, [M,N]) ; bit1: bf16 output, else f32 output.
__global__ void __launch_bounds__(256)
wmma_gemm_bf16(const unsigned short* __restrict__ A,
               const unsigned short* __restrict__ W,
               const float* __restrict__ resid,
               float* __restrict__ outF,
               unsigned short* __restrict__ outB,
               int M, int N, int K, int flags) {
    __shared__ __align__(16) unsigned short Wlds[2 * 64 * 32];   // 8 KB, double buffered

    int tid  = threadIdx.x;            // 0..255
    int wave = tid >> 5;               // 0..7
    int lane = tid & 31;
    int half = lane >> 4;              // 0 | 1
    int l16  = lane & 15;

    int n0   = blockIdx.x * 64;
    int m0   = blockIdx.y * 128;
    int mrow = m0 + wave * 16;         // this wave's 16 output rows

    // async-copy assignment: thread copies 16B: row = tid/4 (0..63), chunk = tid%4
    int crow = tid >> 2;
    int cch  = tid & 3;
    const unsigned short* wsrc = W + (size_t)(n0 + crow) * K + cch * 8;
    unsigned ldsdst0 = (unsigned)(size_t)(const void*)&Wlds[crow * 32 + cch * 8];
    unsigned ldsdst1 = ldsdst0 + 4096;                            // + 2048 halves

    const unsigned short* arow = A + (size_t)(mrow + l16) * K + half * 16;

    v8f acc[4];
    #pragma unroll
    for (int nt = 0; nt < 4; ++nt)
        #pragma unroll
        for (int i = 0; i < 8; ++i) acc[nt][i] = 0.f;

    // one k-step: batched frag loads, then 4 back-to-back WMMAs
    auto step = [&](const unsigned short* wbuf, int kk) {
        FragBF a;
        a.u[0] = *(const u32x4*)(arow + kk);        // K = kbA + 0..7
        a.u[1] = *(const u32x4*)(arow + kk + 8);    // K = kbA + 8..15
        __builtin_prefetch(arow + kk + 256, 0, 1);  // global_prefetch_b8
        const unsigned short* wp = wbuf + l16 * 32 + half * 16;
        FragBF b0, b1, b2, b3;
        b0.u[0] = *(const u32x4*)(wp);              // ds_load_b128 x8, one wait
        b0.u[1] = *(const u32x4*)(wp + 8);
        b1.u[0] = *(const u32x4*)(wp + 16 * 32);
        b1.u[1] = *(const u32x4*)(wp + 16 * 32 + 8);
        b2.u[0] = *(const u32x4*)(wp + 32 * 32);
        b2.u[1] = *(const u32x4*)(wp + 32 * 32 + 8);
        b3.u[0] = *(const u32x4*)(wp + 48 * 32);
        b3.u[1] = *(const u32x4*)(wp + 48 * 32 + 8);
        acc[0] = wmma_bf16(a, b0, acc[0]);
        acc[1] = wmma_bf16(a, b1, acc[1]);
        acc[2] = wmma_bf16(a, b2, acc[2]);
        acc[3] = wmma_bf16(a, b3, acc[3]);
    };

    // prologue: stage k0 = 0 into buffer 0
    async_ld_lds_b128(ldsdst0, wsrc);

    unsigned buf = 0;
    int k0 = 0;
    for (; k0 + 32 < K; k0 += 32) {                 // branch-free body
        async_ld_lds_b128(buf ? ldsdst0 : ldsdst1, wsrc + k0 + 32);
        wait_async_le1();                           // current buffer's DMA done (in-order)
        __syncthreads();                            // tile visible to all 8 waves
        step(buf ? Wlds + 2048 : Wlds, k0);
        __syncthreads();                            // reads done before re-fill
        buf ^= 1;
    }
    wait_async_le0();                               // peeled last k-step
    __syncthreads();
    step(buf ? Wlds + 2048 : Wlds, k0);

    // epilogue
    #pragma unroll
    for (int nt = 0; nt < 4; ++nt) {
        int n = n0 + nt * 16 + l16;
        #pragma unroll
        for (int r = 0; r < 8; ++r) {
            int m = mrow + half * 8 + r;
            float v = acc[nt][r];
            if (flags & 1) v += resid[(size_t)m * N + n];
            if (flags & 2) outB[(size_t)m * N + n] = f32_to_bf16(v);
            else           outF[(size_t)m * N + n] = v;
        }
    }
}

// ---------------------------------------------------------------- RoPE (in-place on bf16 q,k)
__global__ void rope_kernel(unsigned short* __restrict__ q,
                            unsigned short* __restrict__ k) {
    int t   = blockIdx.x;            // token
    int s   = t & (S - 1);
    int tid = threadIdx.x;           // 0..511
    int h   = tid >> 5;
    int i   = tid & 31;
    float inv = __powf(10000.f, -(float)i / 32.f);
    float fr  = (float)s * inv;
    float cs  = __cosf(fr), sn = __sinf(fr);
    size_t base = (size_t)t * D + h * DK;

    float q1 = bf16_to_f32(q[base + i]), q2 = bf16_to_f32(q[base + i + 32]);
    q[base + i]      = f32_to_bf16(q1 * cs - q2 * sn);
    q[base + i + 32] = f32_to_bf16(q2 * cs + q1 * sn);

    float k1 = bf16_to_f32(k[base + i]), k2 = bf16_to_f32(k[base + i + 32]);
    k[base + i]      = f32_to_bf16(k1 * cs - k2 * sn);
    k[base + i + 32] = f32_to_bf16(k2 * cs + k1 * sn);
}

// ---------------------------------------------------------------- flash attention (one wave / 16-row q tile)
__global__ void attention_kernel(const unsigned short* __restrict__ qb,
                                 const unsigned short* __restrict__ kbuf,
                                 const unsigned short* __restrict__ vb,
                                 unsigned short* __restrict__ ctx) {
    constexpr float scale = 0.125f;               // 1/sqrt(64)
    int q0 = blockIdx.x * 16;
    int bi = blockIdx.y / H;
    int hi = blockIdx.y % H;
    size_t tok0 = (size_t)bi * S;

    int lane = threadIdx.x & 31;
    int half = lane >> 4;
    int l16  = lane & 15;

    // q A-frags for dk-chunks 0..31 and 32..63
    FragBF aq0, aq1;
    {
        const unsigned short* qp = qb + (tok0 + q0 + l16) * D + hi * DK + half * 16;
        aq0.u[0] = *(const u32x4*)(qp);
        aq0.u[1] = *(const u32x4*)(qp + 8);
        aq1.u[0] = *(const u32x4*)(qp + 32);
        aq1.u[1] = *(const u32x4*)(qp + 40);
    }

    float m_i[8], l_i[8];
    v8f acc[4];
    #pragma unroll
    for (int r = 0; r < 8; ++r) { m_i[r] = -3e38f; l_i[r] = 0.f; }
    #pragma unroll
    for (int dt = 0; dt < 4; ++dt)
        #pragma unroll
        for (int r = 0; r < 8; ++r) acc[dt][r] = 0.f;

    __shared__ __align__(16) unsigned short Plds[16 * 32];

    for (int kblk = 0; kblk < q0 + 16; kblk += 32) {
        // ---- scores: 16 q-rows x 32 keys, via 2 N-tiles x 2 K-steps
        v8f sv[2];
        #pragma unroll
        for (int nt = 0; nt < 2; ++nt) {
            int kt = kblk + nt * 16;
            const unsigned short* kp = kbuf + (tok0 + kt + l16) * D + hi * DK + half * 16;
            FragBF bk0, bk1;
            bk0.u[0] = *(const u32x4*)(kp);
            bk0.u[1] = *(const u32x4*)(kp + 8);
            bk1.u[0] = *(const u32x4*)(kp + 32);
            bk1.u[1] = *(const u32x4*)(kp + 40);
            v8f s;
            #pragma unroll
            for (int i = 0; i < 8; ++i) s[i] = 0.f;
            s = wmma_bf16(aq0, bk0, s);
            s = wmma_bf16(aq1, bk1, s);
            sv[nt] = s;
        }
        // ---- scale + causal mask + online softmax (row = q0 + half*8 + r, col = kt + l16)
        #pragma unroll
        for (int r = 0; r < 8; ++r) {
            int qi  = q0 + half * 8 + r;
            int k0i = kblk + l16;
            int k1i = kblk + 16 + l16;
            float s0 = sv[0][r] * scale; if (k0i > qi) s0 = -1e9f;
            float s1 = sv[1][r] * scale; if (k1i > qi) s1 = -1e9f;

            float mx = fmaxf(s0, s1);
            #pragma unroll
            for (int off = 8; off >= 1; off >>= 1)
                mx = fmaxf(mx, __shfl_xor(mx, off, 32));
            float mnew  = fmaxf(m_i[r], mx);
            float alpha = __expf(m_i[r] - mnew);
            float p0 = __expf(s0 - mnew);
            float p1 = __expf(s1 - mnew);
            float rs = p0 + p1;
            #pragma unroll
            for (int off = 8; off >= 1; off >>= 1)
                rs += __shfl_xor(rs, off, 32);
            l_i[r] = l_i[r] * alpha + rs;
            m_i[r] = mnew;
            #pragma unroll
            for (int dt = 0; dt < 4; ++dt) acc[dt][r] *= alpha;

            int prow = half * 8 + r;
            Plds[prow * 32 + l16]      = f32_to_bf16(p0);
            Plds[prow * 32 + 16 + l16] = f32_to_bf16(p1);
        }
        __syncthreads();

        // ---- P (16x32) as A-frag from LDS
        FragBF pf;
        {
            const unsigned short* pp = &Plds[l16 * 32 + half * 8];
            pf.u[0] = *(const u32x4*)(pp);
            pf.u[1] = *(const u32x4*)(pp + 16);
        }
        // ---- P @ V, 4 dk tiles of 16: batch all V-frag loads, then 4 WMMAs
        FragBF vf[4];
        #pragma unroll
        for (int dt = 0; dt < 4; ++dt)
            #pragma unroll
            for (int j = 0; j < 16; ++j) {
                int key = kblk + half * 8 + (j & 7) + (j >> 3) * 16;
                vf[dt].h[j] = vb[(tok0 + key) * D + hi * DK + dt * 16 + l16];
            }
        #pragma unroll
        for (int dt = 0; dt < 4; ++dt)
            acc[dt] = wmma_bf16(pf, vf[dt], acc[dt]);
        __syncthreads();
    }

    // ---- epilogue: ctx = acc / l_i  (bf16)
    #pragma unroll
    for (int dt = 0; dt < 4; ++dt)
        #pragma unroll
        for (int r = 0; r < 8; ++r) {
            int m = q0 + half * 8 + r;
            ctx[(tok0 + m) * D + hi * DK + dt * 16 + l16] =
                f32_to_bf16(acc[dt][r] / l_i[r]);
        }
}

// ---------------------------------------------------------------- SwiGLU gate
__global__ void silu_mul_kernel(const float* __restrict__ g1,
                                const float* __restrict__ g3,
                                unsigned short* __restrict__ ff) {
    size_t i = (size_t)blockIdx.x * blockDim.x + threadIdx.x;
    float a  = g1[i];
    float s  = a / (1.f + __expf(-a));
    ff[i] = f32_to_bf16(s * g3[i]);
}

// ---------------------------------------------------------------- host orchestration
extern "C" void kernel_launch(void* const* d_in, const int* in_sizes, int n_in,
                              void* d_out, int out_size, void* d_ws, size_t ws_size,
                              hipStream_t stream) {
    (void)in_sizes; (void)n_in; (void)out_size; (void)ws_size;

    const int*   ids    = (const int*)  d_in[0];
    const float* tok    = (const float*)d_in[1];
    const float* pos    = (const float*)d_in[2];
    const float* wq     = (const float*)d_in[3];
    const float* wk     = (const float*)d_in[4];
    const float* wv     = (const float*)d_in[5];
    const float* wo     = (const float*)d_in[6];
    const float* w1     = (const float*)d_in[7];
    const float* w2     = (const float*)d_in[8];
    const float* w3     = (const float*)d_in[9];
    const float* ln1_g  = (const float*)d_in[10];
    const float* ln1_b  = (const float*)d_in[11];
    const float* ln2_g  = (const float*)d_in[12];
    const float* ln2_b  = (const float*)d_in[13];
    const float* lnf_g  = (const float*)d_in[14];
    const float* lnf_b  = (const float*)d_in[15];
    float*       out    = (float*)d_out;

    // ---- workspace carve-up (256B aligned)
    char*  base = (char*)d_ws;
    size_t off  = 0;
    auto carve = [&](size_t bytes) {
        void* p = base + off;
        off = (off + bytes + 255) & ~(size_t)255;
        return p;
    };
    const size_t nQKVO = (size_t)L * D * D;   // per-tensor elements
    const size_t nFFN  = (size_t)L * F * D;

    unsigned short* wq_bf = (unsigned short*)carve(nQKVO * 2);
    unsigned short* wk_bf = (unsigned short*)carve(nQKVO * 2);
    unsigned short* wv_bf = (unsigned short*)carve(nQKVO * 2);
    unsigned short* wo_bf = (unsigned short*)carve(nQKVO * 2);
    unsigned short* w1_bf = (unsigned short*)carve(nFFN * 2);
    unsigned short* w2_bf = (unsigned short*)carve(nFFN * 2);
    unsigned short* w3_bf = (unsigned short*)carve(nFFN * 2);

    float*          x     = (float*)carve((size_t)T * D * 4);
    unsigned short* h_bf  = (unsigned short*)carve((size_t)T * D * 2);
    unsigned short* q_bf  = (unsigned short*)carve((size_t)T * D * 2);
    unsigned short* k_bf  = (unsigned short*)carve((size_t)T * D * 2);
    unsigned short* v_bf  = (unsigned short*)carve((size_t)T * D * 2);
    unsigned short* c_bf  = (unsigned short*)carve((size_t)T * D * 2);
    float*          g1    = (float*)carve((size_t)T * F * 4);
    float*          g3    = (float*)carve((size_t)T * F * 4);
    unsigned short* ff_bf = (unsigned short*)carve((size_t)T * F * 2);

    // ---- weights -> bf16 (once per launch; deterministic)
    cvt_f32_bf16<<<4096, 256, 0, stream>>>(wq, wq_bf, nQKVO);
    cvt_f32_bf16<<<4096, 256, 0, stream>>>(wk, wk_bf, nQKVO);
    cvt_f32_bf16<<<4096, 256, 0, stream>>>(wv, wv_bf, nQKVO);
    cvt_f32_bf16<<<4096, 256, 0, stream>>>(wo, wo_bf, nQKVO);
    cvt_f32_bf16<<<8192, 256, 0, stream>>>(w1, w1_bf, nFFN);
    cvt_f32_bf16<<<8192, 256, 0, stream>>>(w2, w2_bf, nFFN);
    cvt_f32_bf16<<<8192, 256, 0, stream>>>(w3, w3_bf, nFFN);

    // ---- embedding
    embed_kernel<<<T, 256, 0, stream>>>(ids, tok, pos, x);

    dim3 gD (D / 64, T / 128);   // (16, 16)
    dim3 gF (F / 64, T / 128);   // (64, 16)
    dim3 gAtt(S / 16, B * H);    // (64, 32)

    for (int l = 0; l < L; ++l) {
        const size_t oQ = (size_t)l * D * D;
        const size_t oM = (size_t)l * F * D;

        ln_bf16_kernel<<<T, 256, 0, stream>>>(x, ln1_g + l * D, ln1_b + l * D, h_bf);

        wmma_gemm_bf16<<<gD, 256, 0, stream>>>(h_bf, wq_bf + oQ, nullptr, nullptr, q_bf, T, D, D, 2);
        wmma_gemm_bf16<<<gD, 256, 0, stream>>>(h_bf, wk_bf + oQ, nullptr, nullptr, k_bf, T, D, D, 2);
        wmma_gemm_bf16<<<gD, 256, 0, stream>>>(h_bf, wv_bf + oQ, nullptr, nullptr, v_bf, T, D, D, 2);

        rope_kernel<<<T, 512, 0, stream>>>(q_bf, k_bf);

        attention_kernel<<<gAtt, 32, 0, stream>>>(q_bf, k_bf, v_bf, c_bf);

        // x = x + ctx @ wo^T
        wmma_gemm_bf16<<<gD, 256, 0, stream>>>(c_bf, wo_bf + oQ, x, x, nullptr, T, D, D, 1);

        ln_bf16_kernel<<<T, 256, 0, stream>>>(x, ln2_g + l * D, ln2_b + l * D, h_bf);

        wmma_gemm_bf16<<<gF, 256, 0, stream>>>(h_bf, w1_bf + oM, nullptr, g1, nullptr, T, F, D, 0);
        wmma_gemm_bf16<<<gF, 256, 0, stream>>>(h_bf, w3_bf + oM, nullptr, g3, nullptr, T, F, D, 0);

        silu_mul_kernel<<<(T * (size_t)F) / 256, 256, 0, stream>>>(g1, g3, ff_bf);

        // x = x + ff @ w2^T
        wmma_gemm_bf16<<<gD, 256, 0, stream>>>(ff_bf, w2_bf + oM, x, x, nullptr, T, D, F, 1);
    }

    ln_final_kernel<<<B, 256, 0, stream>>>(x, lnf_g, lnf_b, out);
}
